// Attention_34583076667678
// MI455X (gfx1250) — matrix operand order repaired and checked by reading, not compiled, not run on hardware
//
#include <hip/hip_runtime.h>
#include <hip/hip_bf16.h>

// ---------------------------------------------------------------------------
// ViT attention for MI455X (gfx1250, wave32, WMMA + Tensor Data Mover).
// All matmuls: v_wmma_f32_16x16x32_bf16 (bf16 operands, fp32 accumulate).
// Attention K/V tiles staged LDS via TDM (tensor_load_to_lds), double-buffered.
// ---------------------------------------------------------------------------

typedef __attribute__((ext_vector_type(16))) __bf16 v16bf;
typedef __attribute__((ext_vector_type(8)))  __bf16 v8bf;
typedef __attribute__((ext_vector_type(8)))  float  v8f;

typedef __attribute__((ext_vector_type(4))) unsigned int tdm_v4u;
typedef __attribute__((ext_vector_type(8))) int          tdm_v8i;
typedef __attribute__((ext_vector_type(4))) int          tdm_v4i;

#define NB   32
#define SEQ  577
#define CH   768
#define NH   12
#define DH   64
#define NPAD 640          // SEQ padded to 10 key tiles of 64
#define KTILES (NPAD/64)
#define QKVC (3*CH)       // 2304
#define ROWS (NB*SEQ)     // 18464 = 1154 * 16 exactly
#define ROWT (ROWS/16)    // 1154

static_assert(ROWS % 16 == 0, "row tiles must be exact");

__device__ __forceinline__ v8f wmma_bf16(v16bf a, v16bf b, v8f c) {
  // (neg_a, A, neg_b, B, c_mod, C, reuse_a, reuse_b)
  return __builtin_amdgcn_wmma_f32_16x16x32_bf16(false, a, false, b,
                                                 (short)0, c, false, false);
}

// A-fragment 16x32 (MxK), src row-major, leading dim ld (elements), base
// already at [row0, k0].  Lane L -> row (L&15); K elems {h*16+0..15}.
__device__ __forceinline__ v16bf load_a_frag(const __bf16* base, int ld, int lane) {
  const __bf16* p = base + (size_t)(lane & 15) * ld + ((lane >> 4) << 4);
  v8bf lo = *(const v8bf*)(p);
  v8bf hi = *(const v8bf*)(p + 8);
  v16bf a;
#pragma unroll
  for (int i = 0; i < 8; ++i) { a[i] = lo[i]; a[8 + i] = hi[i]; }
  return a;
}

// B-fragment 32x16 (KxN) with B[k][n] = src[n][k], src row-major, ld elems.
// Lane L -> column n0+(L&15); 16 contiguous K elems at k0 + (L>>4)*16.
__device__ __forceinline__ v16bf load_b_frag(const __bf16* src, int ld,
                                             int n0, int k0, int lane) {
  const __bf16* p = src + (size_t)(n0 + (lane & 15)) * ld + k0 + ((lane >> 4) << 4);
  return *(const v16bf*)p;
}

// ---------------------------------------------------------------------------
// TDM: 2D tile load Global -> LDS.  D# packed per CDNA5 ISA
// §8.3-8.5: group0 {count=1, lds_addr, global_addr, type=2}, group1
// {data_size=1(2B), tensor_dim=tile_dim (no OOB clip), dim0 stride}.
// Wave-level op (EXEC ignored); issue from one wave only.
// 6-arg toolchain form: (v4u g0, v8i g1, v4i g2, v4i g3, v8i aux, i32 cpol).
// ---------------------------------------------------------------------------
__device__ __forceinline__ void tdm_load_2d(const void* gptr, void* lds_ptr,
                                            unsigned tile_d0, unsigned tile_d1,
                                            unsigned stride_elems) {
  unsigned long long ga = (unsigned long long)gptr;
  unsigned lds = (unsigned)(unsigned long long)lds_ptr;   // LDS byte offset
  tdm_v4u g0;
  g0[0] = 1u;                                        // count=1, user descriptor
  g0[1] = lds;                                       // lds_addr
  g0[2] = (unsigned)(ga & 0xffffffffull);            // global_addr[31:0]
  g0[3] = (unsigned)((ga >> 32) & 0x1ffffffull)      // global_addr[56:32]
        | (2u << 30);                                // type = 2 ("image")
  tdm_v8i g1;
  g1[0] = (int)(1u << 16);                           // data_size = 1 -> 2 bytes
  g1[1] = (int)((tile_d0 & 0xffffu) << 16);          // tensor_dim0[15:0]
  g1[2] = (int)(((tile_d0 >> 16) & 0xffffu) |        // tensor_dim0[31:16]
                ((tile_d1 & 0xffffu) << 16));        // tensor_dim1[15:0]
  g1[3] = (int)(((tile_d1 >> 16) & 0xffffu) |        // tensor_dim1[31:16]
                ((tile_d0 & 0xffffu) << 16));        // tile_dim0
  g1[4] = (int)(tile_d1 & 0xffffu);                  // tile_dim1 (tile_dim2=0)
  g1[5] = (int)stride_elems;                         // tensor_dim0_stride[31:0]
  g1[6] = 0;                                         // stride[47:32], dim1_stride
  g1[7] = 0;
  tdm_v4i gz4 = {0, 0, 0, 0};
  tdm_v8i gz8 = {0, 0, 0, 0, 0, 0, 0, 0};
  __builtin_amdgcn_tensor_load_to_lds(g0, g1, gz4, gz4, gz8, 0);
}

// ---------------------------------------------------------------------------
// Kernel 1: fp32 -> bf16, 8 elements/thread (32B load, 16B store)
// ---------------------------------------------------------------------------
__global__ void cvt_bf16_kernel(const float* __restrict__ in,
                                __bf16* __restrict__ out, int n8) {
  int i = blockIdx.x * blockDim.x + threadIdx.x;
  if (i < n8) {
    float4 a = ((const float4*)in)[i * 2 + 0];
    float4 b = ((const float4*)in)[i * 2 + 1];
    v8bf o;
    o[0] = (__bf16)a.x; o[1] = (__bf16)a.y; o[2] = (__bf16)a.z; o[3] = (__bf16)a.w;
    o[4] = (__bf16)b.x; o[5] = (__bf16)b.y; o[6] = (__bf16)b.z; o[7] = (__bf16)b.w;
    ((v8bf*)out)[i] = o;
  }
}

// ---------------------------------------------------------------------------
// Kernel 2: QKV GEMM + RoPE + scale.  Each wave owns a 16-row x 64-col tile:
// one A fragment feeds 4 WMMAs.  Grid: (ceil(ROWT/4), QKVC/64), 128 thr.
// ---------------------------------------------------------------------------
__global__ __launch_bounds__(128) void qkv_rope_kernel(
    const __bf16* __restrict__ xb, const __bf16* __restrict__ wqkv,
    const float* __restrict__ freqs,
    __bf16* __restrict__ qb, __bf16* __restrict__ kb, __bf16* __restrict__ vbT) {
  const int lane = threadIdx.x & 31;
  const int wv   = threadIdx.x >> 5;
  const int rowTile = blockIdx.x * 4 + wv;
  if (rowTile >= ROWT) return;                 // no barriers in this kernel
  const int m0 = rowTile * 16;
  const int j0 = blockIdx.y * 64;

  v8f acc[4] = {v8f{}, v8f{}, v8f{}, v8f{}};
#pragma unroll 2
  for (int k0 = 0; k0 < CH; k0 += 32) {
    v16bf a = load_a_frag(xb + (size_t)m0 * CH + k0, CH, lane);
#pragma unroll
    for (int f = 0; f < 4; ++f)
      acc[f] = wmma_bf16(a, load_b_frag(wqkv, CH, j0 + f * 16, k0, lane), acc[f]);
  }

  const int mat = j0 / CH;                     // 0=q 1=k 2=v (uniform)
  const int h   = (j0 % CH) >> 6;              // uniform (64-col tile)
#pragma unroll
  for (int f = 0; f < 4; ++f) {
    const int d = f * 16 + (lane & 15);        // j0 % 64 == 0
#pragma unroll
    for (int r = 0; r < 8; ++r) {
      int m  = m0 + ((lane >> 4) << 3) + r;    // C-layout row
      int b_ = m / SEQ;
      int n  = m - b_ * SEQ;
      float v = acc[f][r];
      float partner = __shfl_xor(v, 1, 32);    // RoPE pair = adjacent lane
      if (mat < 2 && n >= 1) {
        float ang = freqs[(size_t)(n - 1) * (DH / 2) + (d >> 1)];
        float cs = __cosf(ang), sn = __sinf(ang);
        v = (d & 1) ? (partner * sn + v * cs)  // imag
                    : (v * cs - partner * sn); // real
      }
      if (mat == 0) v *= 0.125f;               // Dh^-0.5 folded into q
      __bf16 bv = (__bf16)v;
      size_t bh = (size_t)b_ * NH + h;
      if (mat == 0)      qb[(bh * NPAD + n) * DH + d] = bv;
      else if (mat == 1) kb[(bh * NPAD + n) * DH + d] = bv;
      else               vbT[(bh * DH + d) * NPAD + n] = bv;
    }
  }
}

// ---------------------------------------------------------------------------
// Kernel 3: flash attention.  Grid (NPAD/64 query tiles, B*H).
// K/V 64x64 tiles -> LDS via TDM, double buffered; P via LDS re-tiling.
// ---------------------------------------------------------------------------
__global__ __launch_bounds__(128) void attn_kernel(
    const __bf16* __restrict__ qb, const __bf16* __restrict__ kb,
    const __bf16* __restrict__ vbT, __bf16* __restrict__ ob) {
  __shared__ __bf16 Kt[2][64 * DH];     // 2 x 8 KB  (rows = key, cols = dh)
  __shared__ __bf16 Vt[2][DH * 64];     // 2 x 8 KB  (rows = dh, cols = key)
  __shared__ __bf16 Pl[4][16 * DH];     // 8 KB, one 16x64 tile per wave

  const int lane = threadIdx.x & 31;
  const int wv   = threadIdx.x >> 5;
  const int bh   = blockIdx.y;
  const int m0   = blockIdx.x * 64 + wv * 16;

  const __bf16* qbase = qb  + (size_t)bh * NPAD * DH;
  const __bf16* kbase = kb  + (size_t)bh * NPAD * DH;
  const __bf16* vbase = vbT + (size_t)bh * DH * NPAD;
  __bf16* Pw = Pl[wv];

  v16bf qa0 = load_a_frag(qbase + (size_t)m0 * DH + 0,  DH, lane);
  v16bf qa1 = load_a_frag(qbase + (size_t)m0 * DH + 32, DH, lane);

  v8f o0 = {}, o1 = {}, o2 = {}, o3 = {};
  float mrow[8], lrow[8];
#pragma unroll
  for (int r = 0; r < 8; ++r) { mrow[r] = -__builtin_inff(); lrow[r] = 0.f; }

  if (wv == 0) {                         // prologue: tile 0 into buffer 0
    tdm_load_2d(kbase, &Kt[0][0], DH, 64, DH);
    tdm_load_2d(vbase, &Vt[0][0], 64, DH, NPAD);
  }

  for (int kt = 0; kt < KTILES; ++kt) {
    const int key0 = kt * 64;
    const int buf  = kt & 1;

    if (wv == 0) {
      if (kt + 1 < KTILES) {             // prefetch next tile into other buffer
        tdm_load_2d(kbase + (size_t)(key0 + 64) * DH, &Kt[buf ^ 1][0], DH, 64, DH);
        tdm_load_2d(vbase + key0 + 64,               &Vt[buf ^ 1][0], 64, DH, NPAD);
        __builtin_amdgcn_s_wait_tensorcnt(2);   // current tile's 2 TDM ops done
      } else {
        __builtin_amdgcn_s_wait_tensorcnt(0);
      }
    }
    __syncthreads();                     // tile (kt) visible to all waves

    // ---- S = Q * K^T  (16 x 64), K from LDS ----
    v8f s[4];
#pragma unroll
    for (int f = 0; f < 4; ++f) {
      s[f] = (v8f){};
      s[f] = wmma_bf16(qa0, load_b_frag(&Kt[buf][0], DH, f * 16, 0,  lane), s[f]);
      s[f] = wmma_bf16(qa1, load_b_frag(&Kt[buf][0], DH, f * 16, 32, lane), s[f]);
    }

    // ---- mask padded keys ----
#pragma unroll
    for (int f = 0; f < 4; ++f) {
      if (key0 + f * 16 + (lane & 15) >= SEQ) {
#pragma unroll
        for (int r = 0; r < 8; ++r) s[f][r] = -1e30f;
      }
    }

    // ---- online softmax ----
    float mnew[8], corr[8];
#pragma unroll
    for (int r = 0; r < 8; ++r) {
      float mx = fmaxf(fmaxf(s[0][r], s[1][r]), fmaxf(s[2][r], s[3][r]));
#pragma unroll
      for (int off = 1; off <= 8; off <<= 1)
        mx = fmaxf(mx, __shfl_xor(mx, off, 32));
      mnew[r] = fmaxf(mrow[r], mx);
      corr[r] = __expf(mrow[r] - mnew[r]);
      mrow[r] = mnew[r];
    }
#pragma unroll
    for (int f = 0; f < 4; ++f)
#pragma unroll
      for (int r = 0; r < 8; ++r) s[f][r] = __expf(s[f][r] - mnew[r]);
#pragma unroll
    for (int r = 0; r < 8; ++r) {
      float t = s[0][r] + s[1][r] + s[2][r] + s[3][r];
#pragma unroll
      for (int off = 1; off <= 8; off <<= 1) t += __shfl_xor(t, off, 32);
      lrow[r] = lrow[r] * corr[r] + t;
      o0[r] *= corr[r]; o1[r] *= corr[r]; o2[r] *= corr[r]; o3[r] *= corr[r];
    }

    // ---- P: C-layout -> LDS (row-major 16x64) ----
#pragma unroll
    for (int f = 0; f < 4; ++f)
#pragma unroll
      for (int r = 0; r < 8; ++r)
        Pw[(size_t)(((lane >> 4) << 3) + r) * DH + f * 16 + (lane & 15)] =
            (__bf16)s[f][r];
    __syncthreads();

    // ---- O += P * V, V from LDS (transposed: contiguous along keys) ----
#pragma unroll
    for (int kk = 0; kk < 64; kk += 32) {
      v16bf ap = load_a_frag(Pw + kk, DH, lane);
      o0 = wmma_bf16(ap, load_b_frag(&Vt[buf][0], 64,  0, kk, lane), o0);
      o1 = wmma_bf16(ap, load_b_frag(&Vt[buf][0], 64, 16, kk, lane), o1);
      o2 = wmma_bf16(ap, load_b_frag(&Vt[buf][0], 64, 32, kk, lane), o2);
      o3 = wmma_bf16(ap, load_b_frag(&Vt[buf][0], 64, 48, kk, lane), o3);
    }
    __syncthreads();                     // done with buffer kt before overwrite
  }

  // ---- epilogue: normalize, store (B,N,C) bf16 ----
  const int b_ = bh / NH, h = bh % NH;
#pragma unroll
  for (int r = 0; r < 8; ++r) {
    int n = m0 + ((lane >> 4) << 3) + r;
    if (n < SEQ) {
      float inv = 1.0f / lrow[r];
      size_t rowbase = ((size_t)b_ * SEQ + n) * CH + h * DH + (lane & 15);
      ob[rowbase +  0] = (__bf16)(o0[r] * inv);
      ob[rowbase + 16] = (__bf16)(o1[r] * inv);
      ob[rowbase + 32] = (__bf16)(o2[r] * inv);
      ob[rowbase + 48] = (__bf16)(o3[r] * inv);
    }
  }
}

// ---------------------------------------------------------------------------
// Kernel 4: output projection + bias, fp32 to d_out.  Wave = 16x64 tile.
// ---------------------------------------------------------------------------
__global__ __launch_bounds__(128) void proj_kernel(
    const __bf16* __restrict__ ob, const __bf16* __restrict__ wproj,
    const float* __restrict__ bias, float* __restrict__ out) {
  const int lane = threadIdx.x & 31;
  const int wv   = threadIdx.x >> 5;
  const int rowTile = blockIdx.x * 4 + wv;
  if (rowTile >= ROWT) return;
  const int m0 = rowTile * 16;
  const int j0 = blockIdx.y * 64;

  v8f acc[4] = {v8f{}, v8f{}, v8f{}, v8f{}};
#pragma unroll 2
  for (int k0 = 0; k0 < CH; k0 += 32) {
    v16bf a = load_a_frag(ob + (size_t)m0 * CH + k0, CH, lane);
#pragma unroll
    for (int f = 0; f < 4; ++f)
      acc[f] = wmma_bf16(a, load_b_frag(wproj, CH, j0 + f * 16, k0, lane), acc[f]);
  }
#pragma unroll
  for (int f = 0; f < 4; ++f) {
    const int j = j0 + f * 16 + (lane & 15);
    const float bj = bias[j];
#pragma unroll
    for (int r = 0; r < 8; ++r) {
      int m = m0 + ((lane >> 4) << 3) + r;
      out[(size_t)m * CH + j] = acc[f][r] + bj;
    }
  }
}

// ---------------------------------------------------------------------------
extern "C" void kernel_launch(void* const* d_in, const int* in_sizes, int n_in,
                              void* d_out, int out_size, void* d_ws, size_t ws_size,
                              hipStream_t stream) {
  const float* x     = (const float*)d_in[0];
  const float* w_qkv = (const float*)d_in[1];
  const float* w_prj = (const float*)d_in[2];
  const float* b_prj = (const float*)d_in[3];
  const float* freqs = (const float*)d_in[4];

  char* w = (char*)d_ws;
  size_t off = 0;
  auto carve = [&](size_t elems) {
    void* p = w + off;
    off += (elems * sizeof(unsigned short) + 255) & ~(size_t)255;
    return (__bf16*)p;
  };
  __bf16* xb    = carve((size_t)ROWS * CH);
  __bf16* wqkvb = carve((size_t)QKVC * CH);
  __bf16* wprjb = carve((size_t)CH * CH);
  __bf16* qb    = carve((size_t)NB * NH * NPAD * DH);
  __bf16* kb    = carve((size_t)NB * NH * NPAD * DH);
  __bf16* vbT   = carve((size_t)NB * NH * DH * NPAD);
  __bf16* ob    = carve((size_t)ROWS * CH);

  const int n_x = ROWS * CH, n_wq = QKVC * CH, n_wp = CH * CH;
  cvt_bf16_kernel<<<(n_x / 8 + 255) / 256, 256, 0, stream>>>(x,     xb,    n_x / 8);
  cvt_bf16_kernel<<<(n_wq / 8 + 255) / 256, 256, 0, stream>>>(w_qkv, wqkvb, n_wq / 8);
  cvt_bf16_kernel<<<(n_wp / 8 + 255) / 256, 256, 0, stream>>>(w_prj, wprjb, n_wp / 8);

  dim3 g2((ROWT + 3) / 4, QKVC / 64);
  qkv_rope_kernel<<<g2, 128, 0, stream>>>(xb, wqkvb, freqs, qb, kb, vbT);

  dim3 g3(NPAD / 64, NB * NH);
  attn_kernel<<<g3, 128, 0, stream>>>(qb, kb, vbT, ob);

  dim3 g4((ROWT + 3) / 4, CH / 64);
  proj_kernel<<<g4, 128, 0, stream>>>(ob, wprjb, b_prj, (float*)d_out);
}
